// SelfAttention_60172491817302
// MI455X (gfx1250) — compile-verified, matrix operand order checked by reading, not run on hardware
//
#include <hip/hip_runtime.h>

// ---------------------------------------------------------------------------
// Self-attention for MI455X (gfx1250): bf16 WMMA + async-LDS double buffering.
// B=4, S=4096, D_IN=1024, D_K=128. fp32 in/out, bf16 WMMA with f32 accum.
// ---------------------------------------------------------------------------

typedef __attribute__((ext_vector_type(16))) __bf16 v16bf;
typedef __attribute__((ext_vector_type(8)))  float  v8f;

#define S_LEN 4096
#define D_IN  1024
#define D_K   128
#define NBAT  4
#define NROWS (NBAT * S_LEN)
#define SCALE 0.08838834764831845f   // 1/sqrt(128)

// float -> bf16, round to nearest even (bit trick)
static __device__ __forceinline__ unsigned short f2bf(float f) {
    union { float f; unsigned u; } a; a.f = f;
    unsigned r = a.u + 0x7FFFu + ((a.u >> 16) & 1u);
    return (unsigned short)(r >> 16);
}
static __device__ __forceinline__ unsigned packbf(float x, float y) {
    return (unsigned)f2bf(x) | ((unsigned)f2bf(y) << 16);
}
static __device__ __forceinline__ v16bf mkfrag(uint4 lo, uint4 hi) {
    union { uint4 u[2]; v16bf v; } t; t.u[0] = lo; t.u[1] = hi; return t.v;
}
static __device__ __forceinline__ uint4 ld128(const unsigned short* p) {
    return *(const uint4*)p;
}

// low 32 bits of a flat pointer to __shared__ == workgroup-relative LDS offset
static __device__ __forceinline__ unsigned lds_off(const void* p) {
    return (unsigned)(size_t)p;
}
// CDNA5 async copy global -> LDS (b128 per lane), tracked by ASYNCcnt
static __device__ __forceinline__ void async_ld_b128(unsigned lds_byte_off,
                                                     const void* gptr) {
    asm volatile("global_load_async_to_lds_b128 %0, %1, off"
                 :: "v"(lds_byte_off),
                    "v"((unsigned long long)(size_t)gptr)
                 : "memory");
}
static __device__ __forceinline__ void wait_async0() {
#if __has_builtin(__builtin_amdgcn_s_wait_asynccnt)
    __builtin_amdgcn_s_wait_asynccnt(0);
#else
    asm volatile("s_wait_asynccnt 0x0" ::: "memory");
#endif
}

// ---------------------------------------------------------------------------
// Kernel 0: W_{q,k,v} [D_IN,D_K] f32 -> Wt bf16 [3][D_K][D_IN] (transposed)
// ---------------------------------------------------------------------------
__global__ void wconv_kernel(const float* __restrict__ Wq,
                             const float* __restrict__ Wk,
                             const float* __restrict__ Wv,
                             unsigned short* __restrict__ Wt) {
    int idx = blockIdx.x * 256 + threadIdx.x;   // 0 .. 3*131072-1
    int m   = idx >> 17;
    int r   = idx & 131071;
    int k   = r >> 7;
    int n   = r & 127;
    const float* W = (m == 0) ? Wq : (m == 1) ? Wk : Wv;
    Wt[(size_t)m * (D_K * D_IN) + (size_t)n * D_IN + k] = f2bf(W[(size_t)k * D_K + n]);
}

// ---------------------------------------------------------------------------
// Kernel 1: QKV projection. grid=(NROWS/128, 3), block=256 (8 waves).
// W k-slices double-buffered in LDS via async copies; V written transposed.
// ---------------------------------------------------------------------------
__global__ void qkv_kernel(const float* __restrict__ x,
                           const unsigned short* __restrict__ Wt,
                           const float* __restrict__ bq,
                           const float* __restrict__ bk,
                           const float* __restrict__ bv,
                           unsigned short* __restrict__ Qb,
                           unsigned short* __restrict__ Kb,
                           unsigned short* __restrict__ Vb) {
    __shared__ __align__(16) unsigned short Wl[2 * 128 * 32];   // 16 KB, 2 slices

    const int m = blockIdx.y;
    const unsigned short* W = Wt + (size_t)m * (D_K * D_IN);
    const float* bias       = (m == 0) ? bq : (m == 1) ? bk : bv;

    const int tid  = threadIdx.x;
    const int lane = tid & 31;
    const int wave = tid >> 5;
    const int l16  = lane & 15;
    const int hoff = (lane >> 4) * 8;

    const int rbase = blockIdx.x * 128 + wave * 16;
    const float* xr = x + (size_t)(rbase + l16) * D_IN;

    // async-stage one 32xD_K W-slice (bf16, [n][32] layout) into buffer `buf`
    auto stage_w = [&](int buf, int kb) {
        #pragma unroll
        for (int i = 0; i < 2; ++i) {
            int idx = tid + 256 * i;          // 0..511 over 512 b128 chunks
            int n   = idx >> 2;               // 0..127
            int c8  = (idx & 3) * 8;          // 0,8,16,24
            async_ld_b128(lds_off(Wl + buf * 4096 + n * 32 + c8),
                          W + (size_t)n * D_IN + kb + c8);
        }
    };

    v8f acc[8];
#pragma unroll
    for (int t = 0; t < 8; ++t)
        acc[t] = (v8f){0.f,0.f,0.f,0.f,0.f,0.f,0.f,0.f};

    int p = 0;
    stage_w(0, 0);
    wait_async0();
    __syncthreads();

    for (int kb = 0; kb < D_IN; kb += 32) {
        if (kb + 32 < D_IN) stage_w(p ^ 1, kb + 32);

        // A fragment: 16 f32 of this lane's row, converted to bf16
        __builtin_prefetch(xr + kb + 64, 0, 1);
        float4 f0 = *(const float4*)(xr + kb + hoff);
        float4 f1 = *(const float4*)(xr + kb + hoff + 4);
        float4 f2 = *(const float4*)(xr + kb + 16 + hoff);
        float4 f3 = *(const float4*)(xr + kb + 16 + hoff + 4);
        uint4 alo, ahi;
        alo.x = packbf(f0.x, f0.y); alo.y = packbf(f0.z, f0.w);
        alo.z = packbf(f1.x, f1.y); alo.w = packbf(f1.z, f1.w);
        ahi.x = packbf(f2.x, f2.y); ahi.y = packbf(f2.z, f2.w);
        ahi.z = packbf(f3.x, f3.y); ahi.w = packbf(f3.z, f3.w);
        v16bf af = mkfrag(alo, ahi);

        // stage all 8 B fragments into registers, then burst 8 WMMAs
        const unsigned short* wb = Wl + p * 4096;
        v16bf bfr[8];
#pragma unroll
        for (int t = 0; t < 8; ++t) {
            const unsigned short* wr = wb + (t * 16 + l16) * 32;
            bfr[t] = mkfrag(ld128(wr + hoff), ld128(wr + 16 + hoff));
        }
#pragma unroll
        for (int t = 0; t < 8; ++t)
            acc[t] = __builtin_amdgcn_wmma_f32_16x16x32_bf16(
                false, af, false, bfr[t], (short)0, acc[t], false, false);

        wait_async0();
        __syncthreads();
        p ^= 1;
    }

    // bias add + store bf16.  Q,K row-major [row][col]; V transposed [b][dim][s]
#pragma unroll
    for (int t = 0; t < 8; ++t) {
        const int col = t * 16 + l16;
        const float bval = bias[col];
#pragma unroll
        for (int r = 0; r < 8; ++r) {
            const int row = rbase + hoff + r;        // flattened b*S + s
            const unsigned short v = f2bf(acc[t][r] + bval);
            if (m == 0) {
                Qb[(size_t)row * D_K + col] = v;
            } else if (m == 1) {
                Kb[(size_t)row * D_K + col] = v;
            } else {
                const int b = row >> 12;             // /4096
                const int s = row & 4095;
                Vb[((size_t)(b * D_K + col)) * S_LEN + s] = v;
            }
        }
    }
}

// ---------------------------------------------------------------------------
// Kernel 2: flash attention. grid=(S/128, B), block=256 (8 waves).
// K and (pre-transposed) V tiles double-buffered in LDS via async copies.
// LDS: 2x K (32KB) + 2x V (32KB) + P (32KB) = 160 KB (< 320 KB/WGP).
// ---------------------------------------------------------------------------
__global__ void attn_kernel(const unsigned short* __restrict__ Qb,
                            const unsigned short* __restrict__ Kb,
                            const unsigned short* __restrict__ Vb,
                            float* __restrict__ out) {
    __shared__ __align__(16) unsigned short smem[5 * 128 * 128];
    // layout (u16 elems): K0 @0, K1 @16384, V0 @32768, V1 @49152, P @65536

    const int b    = blockIdx.y;
    const int qb0  = blockIdx.x * 128;
    const int tid  = threadIdx.x;
    const int lane = tid & 31;
    const int wave = tid >> 5;
    const int l16  = lane & 15;
    const int hoff = (lane >> 4) * 8;

    const unsigned short* Qg = Qb + (size_t)b * S_LEN * D_K;
    const unsigned short* Kg = Kb + (size_t)b * S_LEN * D_K;
    const unsigned short* Vg = Vb + (size_t)b * D_K * S_LEN;   // [dim][key]

    // async-stage one key tile: K [key][dim] straight, V [dim][key] straight
    auto stage = [&](int buf, int j0) {
        unsigned short* kd = smem + buf * 16384;
        unsigned short* vd = smem + 2 * 16384 + buf * 16384;
        #pragma unroll
        for (int i = 0; i < 8; ++i) {
            int idx = tid + 256 * i;      // 0..2047 b128 chunks per tile
            int r   = idx >> 4;           // row (key for K, dim for V)
            int c8  = (idx & 15) * 8;     // u16 column offset
            async_ld_b128(lds_off(kd + r * 128 + c8),
                          Kg + (size_t)(j0 + r) * D_K + c8);
            async_ld_b128(lds_off(vd + r * 128 + c8),
                          Vg + (size_t)r * S_LEN + j0 + c8);
        }
    };

    // Load this wave's 16x128 Q tile into 4 A-fragments (kept in registers)
    const int qrow = qb0 + wave * 16 + l16;
    v16bf qf[4];
#pragma unroll
    for (int c = 0; c < 4; ++c) {
        const unsigned short* ptr = Qg + (size_t)qrow * D_K + c * 32;
        qf[c] = mkfrag(ld128(ptr + hoff), ld128(ptr + 16 + hoff));
    }

    float mrow[8], lrow[8];
    v8f oacc[8];
#pragma unroll
    for (int r = 0; r < 8; ++r) { mrow[r] = -3.0e38f; lrow[r] = 0.f; }
#pragma unroll
    for (int t = 0; t < 8; ++t)
        oacc[t] = (v8f){0.f,0.f,0.f,0.f,0.f,0.f,0.f,0.f};

    int p = 0;
    stage(0, 0);
    wait_async0();
    __syncthreads();

    for (int j0 = 0; j0 < S_LEN; j0 += 128) {
        if (j0 + 128 < S_LEN) stage(p ^ 1, j0 + 128);

        const unsigned short* Kt = smem + p * 16384;
        const unsigned short* Vt = smem + 2 * 16384 + p * 16384;
        unsigned short* Pl       = smem + 4 * 16384;

        // ---- scores: 16x128 = Q x K^T, 32 WMMAs ----
        v8f sacc[8];
#pragma unroll
        for (int jt = 0; jt < 8; ++jt)
            sacc[jt] = (v8f){0.f,0.f,0.f,0.f,0.f,0.f,0.f,0.f};
#pragma unroll
        for (int jt = 0; jt < 8; ++jt) {
            v16bf kf[4];
#pragma unroll
            for (int c = 0; c < 4; ++c) {
                const unsigned short* kr = Kt + (jt * 16 + l16) * 128 + c * 32;
                kf[c] = mkfrag(ld128(kr + hoff), ld128(kr + 16 + hoff));
            }
#pragma unroll
            for (int c = 0; c < 4; ++c)
                sacc[jt] = __builtin_amdgcn_wmma_f32_16x16x32_bf16(
                    false, qf[c], false, kf[c], (short)0, sacc[jt], false, false);
        }

        // ---- online softmax (rows live across 16-lane groups) ----
#pragma unroll
        for (int r = 0; r < 8; ++r) {
            float mx = -3.0e38f;
#pragma unroll
            for (int jt = 0; jt < 8; ++jt) {
                float s = sacc[jt][r] * SCALE;
                sacc[jt][r] = s;
                mx = fmaxf(mx, s);
            }
#pragma unroll
            for (int o = 1; o < 16; o <<= 1)
                mx = fmaxf(mx, __shfl_xor(mx, o, 32));
            const float mnew  = fmaxf(mrow[r], mx);
            const float alpha = __expf(mrow[r] - mnew);
            mrow[r] = mnew;
            float rs = 0.f;
#pragma unroll
            for (int jt = 0; jt < 8; ++jt) {
                float pr = __expf(sacc[jt][r] - mnew);
                sacc[jt][r] = pr;
                rs += pr;
            }
#pragma unroll
            for (int o = 1; o < 16; o <<= 1)
                rs += __shfl_xor(rs, o, 32);
            lrow[r] = lrow[r] * alpha + rs;
#pragma unroll
            for (int dt = 0; dt < 8; ++dt)
                oacc[dt][r] *= alpha;
            const int prow = hoff + r;
#pragma unroll
            for (int jt = 0; jt < 8; ++jt)
                Pl[wave * 2048 + prow * 128 + jt * 16 + l16] = f2bf(sacc[jt][r]);
        }

        // ---- P x V: 32 WMMAs ----
        v16bf pf[4];
#pragma unroll
        for (int c = 0; c < 4; ++c) {
            const unsigned short* pp = Pl + wave * 2048 + l16 * 128 + c * 32;
            pf[c] = mkfrag(ld128(pp + hoff), ld128(pp + 16 + hoff));
        }
#pragma unroll
        for (int dt = 0; dt < 8; ++dt) {
            v16bf vf[4];
#pragma unroll
            for (int c = 0; c < 4; ++c) {
                const unsigned short* vr = Vt + (dt * 16 + l16) * 128 + c * 32;
                vf[c] = mkfrag(ld128(vr + hoff), ld128(vr + 16 + hoff));
            }
#pragma unroll
            for (int c = 0; c < 4; ++c)
                oacc[dt] = __builtin_amdgcn_wmma_f32_16x16x32_bf16(
                    false, pf[c], false, vf[c], (short)0, oacc[dt], false, false);
        }

        wait_async0();
        __syncthreads();
        p ^= 1;
    }

    // ---- normalize and store fp32 output ----
    float* og = out + ((size_t)b * S_LEN + qb0 + wave * 16) * D_K;
#pragma unroll
    for (int r = 0; r < 8; ++r) {
        const float inv = 1.0f / lrow[r];
#pragma unroll
        for (int dt = 0; dt < 8; ++dt)
            og[(hoff + r) * D_K + dt * 16 + l16] = oacc[dt][r] * inv;
    }
}

// ---------------------------------------------------------------------------
extern "C" void kernel_launch(void* const* d_in, const int* in_sizes, int n_in,
                              void* d_out, int out_size, void* d_ws, size_t ws_size,
                              hipStream_t stream) {
    const float* x  = (const float*)d_in[0];
    const float* Wq = (const float*)d_in[1];
    const float* bq = (const float*)d_in[2];
    const float* Wk = (const float*)d_in[3];
    const float* bk = (const float*)d_in[4];
    const float* Wv = (const float*)d_in[5];
    const float* bv = (const float*)d_in[6];

    // workspace (bf16 as u16): Qb | Kb | Vb(transposed) | Wt
    const size_t nQKV = (size_t)NROWS * D_K;
    unsigned short* Qb = (unsigned short*)d_ws;
    unsigned short* Kb = Qb + nQKV;
    unsigned short* Vb = Kb + nQKV;
    unsigned short* Wt = Vb + nQKV;

    wconv_kernel<<<1536, 256, 0, stream>>>(Wq, Wk, Wv, Wt);
    qkv_kernel<<<dim3(NROWS / 128, 3), 256, 0, stream>>>(
        x, Wt, bq, bk, bv, Qb, Kb, Vb);
    attn_kernel<<<dim3(S_LEN / 128, NBAT), 256, 0, stream>>>(
        Qb, Kb, Vb, (float*)d_out);
}
